// SparseVoxelTopDownConvNet_76948634075585
// MI455X (gfx1250) — matrix operand order repaired and checked by reading, not compiled, not run on hardware
//
#include <hip/hip_runtime.h>

typedef __attribute__((ext_vector_type(16))) _Float16 v16h;
typedef __attribute__((ext_vector_type(8)))  _Float16 h8;
typedef __attribute__((ext_vector_type(8)))  float    v8f;

#define KTAPS 27

// ---------------- prep kernels ----------------

__global__ void cvt_f32_f16(const float* __restrict__ in, _Float16* __restrict__ out, int n) {
  int i = blockIdx.x * blockDim.x + threadIdx.x;
  if (i < n) out[i] = (_Float16)in[i];
}

// W [27][cin][cout] f32  ->  Wt [27][cout][cin] f16
__global__ void transpose_w(const float* __restrict__ W, _Float16* __restrict__ Wt,
                            int cin, int cout) {
  int i = blockIdx.x * blockDim.x + threadIdx.x;
  int per = cin * cout;
  int total = KTAPS * per;
  if (i < total) {
    int k = i / per;
    int rem = i - k * per;
    int c = rem / cout;
    int o = rem - c * cout;
    Wt[(size_t)k * per + (size_t)o * cin + c] = (_Float16)W[i];
  }
}

// B fragment (32x16 f16): lane l -> N=l%16; halves 0..7 = K(kc*32 + 8*(l/16) ..), halves 8..15 = K(kc*32 + 16 + 8*(l/16) ..)
template <int LROW>
__device__ __forceinline__ v16h loadB(const _Float16* cbuf, int t, int kc, int ln16, int g) {
  const _Float16* wp = cbuf + (size_t)(t * 16 + ln16) * LROW + kc * 32 + 8 * g;
  h8 b0 = *(const h8*)wp;
  h8 b1 = *(const h8*)(wp + 16);
  v16h b;
#pragma unroll
  for (int i = 0; i < 8; i++) { b[i] = b0[i]; b[8 + i] = b1[i]; }
  return b;
}

// A fragment (16x32 f16): lane l -> M=l%16; halves 0..7 = K(kc*32 + g*8 ..),
// halves 8..15 = K(kc*32 + 16 + g*8 ..) per ISA 16-bit A layout.
__device__ __forceinline__ v16h loadA(const _Float16* xr, bool valid, int kc, int g) {
  v16h a;
#pragma unroll
  for (int i = 0; i < 16; i++) a[i] = (_Float16)0.f;
  if (valid) {
    h8 lo = *(const h8*)(xr + kc * 32 + g * 8);
    h8 hi = *(const h8*)(xr + kc * 32 + 16 + g * 8);
#pragma unroll
    for (int i = 0; i < 8; i++) { a[i] = lo[i]; a[8 + i] = hi[i]; }
  }
  return a;
}

// ---------------- sparse conv kernel ----------------
// One wave -> 32 output rows x COUT cols (two M-tiles share every B fragment).
// Block = 256 threads = 8 waves = 256 rows.  Weights double-buffered in LDS:
// global loads for tap k+1 issue before the tap-k WMMAs, ds_stores land after,
// one barrier per tap.  B fragments are software-pipelined one tile ahead.
// flags: bit0 = lrelu, bit1 = layernorm(gamma,beta), bit2 = f32 output
template <int CIN, int COUT>
__global__ __launch_bounds__(256)
void spconv_kernel(const _Float16* __restrict__ X, int ldx,
                   const int* __restrict__ nbr,               // [Nout][27]
                   const _Float16* __restrict__ Wt,           // [27][COUT][CIN] f16
                   const float* __restrict__ bias,
                   const float* __restrict__ gamma,
                   const float* __restrict__ beta,
                   _Float16* __restrict__ outH,
                   float* __restrict__ outF,
                   int ldo, int colOff, int flags) {
  constexpr int NT   = COUT / 16;        // 16-col tiles
  constexpr int NKC  = CIN / 32;         // 32-deep K chunks
  constexpr int PAD  = 8;                // halves of padding -> conflict-free ds_load_b128
  constexpr int LROW = CIN + PAD;
  constexpr int PT   = (COUT * CIN) / 2048;  // h8 staging chunks per thread
  extern __shared__ _Float16 wlds[];     // [2][COUT][CIN+PAD]

  const int lane  = threadIdx.x & 31;
  const int wave  = threadIdx.x >> 5;
  const int g     = lane >> 4;           // lane-group (ISA half-wave split)
  const int ln16  = lane & 15;
  const long base = (long)blockIdx.x * 256 + wave * 32;
  const long row0 = base + ln16;         // rows gathered by this lane
  const long row1 = base + 16 + ln16;

  v8f acc0[NT], acc1[NT];
#pragma unroll
  for (int t = 0; t < NT; t++) {
    float bc = bias[t * 16 + ln16];
#pragma unroll
    for (int r = 0; r < 8; r++) { acc0[t][r] = bc; acc1[t][r] = bc; }
  }

  // prologue: stage W[0] into buffer 0
  {
    const _Float16* wk = Wt;
#pragma unroll
    for (int p = 0; p < PT; p++) {
      int i = threadIdx.x + p * 256;
      *(h8*)(wlds + (i / (CIN / 8)) * LROW + (i % (CIN / 8)) * 8) =
          *(const h8*)(wk + (size_t)i * 8);
    }
  }

  int idx0 = nbr[row0 * KTAPS];  // tap-0 neighbors, rolled one tap ahead below
  int idx1 = nbr[row1 * KTAPS];

  for (int k = 0; k < KTAPS; k++) {
    __syncthreads();  // W[k] stores visible; all reads of buffer (k+1)&1 retired

    // issue global loads for next tap's weights (latency hidden under WMMAs)
    h8 stg[PT > 0 ? PT : 1];
    int nidx0 = 0, nidx1 = 0;
    if (k + 1 < KTAPS) {
      const _Float16* wk = Wt + (size_t)(k + 1) * COUT * CIN;
#pragma unroll
      for (int p = 0; p < PT; p++)
        stg[p] = *(const h8*)(wk + (size_t)(threadIdx.x + p * 256) * 8);
      nidx0 = nbr[row0 * KTAPS + k + 1];
      nidx1 = nbr[row1 * KTAPS + k + 1];
    }
    if (k + 2 < KTAPS) {  // global_prefetch_b8 for the tap after that
      const char* nw = (const char*)(Wt + (size_t)(k + 2) * COUT * CIN);
      __builtin_prefetch(nw + (size_t)threadIdx.x * 128, 0, 0);
    }

    const _Float16* cbuf = wlds + (size_t)(k & 1) * COUT * LROW;
    const _Float16* xr0 = X + (size_t)(idx0 < 0 ? 0 : idx0) * ldx;
    const _Float16* xr1 = X + (size_t)(idx1 < 0 ? 0 : idx1) * ldx;

#pragma unroll
    for (int kc = 0; kc < NKC; kc++) {
      v16h a0 = loadA(xr0, idx0 >= 0, kc, g);
      v16h a1 = loadA(xr1, idx1 >= 0, kc, g);
      // pipelined B: tile t+1 loads issue before tile t's WMMA pair consumes bcur
      v16h bcur = loadB<LROW>(cbuf, 0, kc, ln16, g);
#pragma unroll
      for (int t = 0; t < NT; t++) {
        v16h bnxt;
        if (t + 1 < NT) bnxt = loadB<LROW>(cbuf, t + 1, kc, ln16, g);
        acc0[t] = __builtin_amdgcn_wmma_f32_16x16x32_f16(
            false, a0, false, bcur, (short)0, acc0[t], false, false);
        acc1[t] = __builtin_amdgcn_wmma_f32_16x16x32_f16(
            false, a1, false, bcur, (short)0, acc1[t], false, false);
        if (t + 1 < NT) bcur = bnxt;
      }
    }

    // store next tap's weights into the other buffer
    if (k + 1 < KTAPS) {
      _Float16* dbuf = wlds + (size_t)((k + 1) & 1) * COUT * LROW;
#pragma unroll
      for (int p = 0; p < PT; p++) {
        int i = threadIdx.x + p * 256;
        *(h8*)(dbuf + (i / (CIN / 8)) * LROW + (i % (CIN / 8)) * 8) = stg[p];
      }
    }
    idx0 = nidx0;
    idx1 = nidx1;
  }

  float gv[NT], bv[NT];
  if (flags & 2) {
#pragma unroll
    for (int t = 0; t < NT; t++) { gv[t] = gamma[t * 16 + ln16]; bv[t] = beta[t * 16 + ln16]; }
  }

#pragma unroll
  for (int sH = 0; sH < 2; sH++) {
    v8f (&A)[NT] = (sH == 0) ? acc0 : acc1;

    if (flags & 1) {  // leaky relu
#pragma unroll
      for (int t = 0; t < NT; t++)
#pragma unroll
        for (int r = 0; r < 8; r++) {
          float v = A[t][r];
          A[t][r] = v >= 0.f ? v : 0.2f * v;
        }
    }

    if (flags & 2) {  // layernorm over COUT channels; rows live in 16-lane halves
#pragma unroll
      for (int r = 0; r < 8; r++) {
        float s = 0.f;
#pragma unroll
        for (int t = 0; t < NT; t++) s += A[t][r];
        s += __shfl_xor(s, 1, 32); s += __shfl_xor(s, 2, 32);
        s += __shfl_xor(s, 4, 32); s += __shfl_xor(s, 8, 32);
        float mu = s * (1.f / COUT);
        float q = 0.f;
#pragma unroll
        for (int t = 0; t < NT; t++) { float d = A[t][r] - mu; q += d * d; }
        q += __shfl_xor(q, 1, 32); q += __shfl_xor(q, 2, 32);
        q += __shfl_xor(q, 4, 32); q += __shfl_xor(q, 8, 32);
        float rs = rsqrtf(q * (1.f / COUT) + 1e-5f);
#pragma unroll
        for (int t = 0; t < NT; t++) A[t][r] = (A[t][r] - mu) * rs * gv[t] + bv[t];
      }
    }

    // D layout: lane l holds (M = r + 8*(l/16), N = t*16 + l%16) in A[t][r]
#pragma unroll
    for (int r = 0; r < 8; r++) {
      long orow = base + sH * 16 + r + 8 * g;
#pragma unroll
      for (int t = 0; t < NT; t++) {
        long o = orow * (long)ldo + colOff + t * 16 + ln16;
        if (flags & 4) outF[o] = A[t][r];
        else           outH[o] = (_Float16)A[t][r];
      }
    }
  }
}

// ---------------- host side ----------------

template <int CIN, int COUT>
static inline size_t shmem_bytes() { return (size_t)2 * COUT * (CIN + 8) * 2; }

static void launch_conv(int cin, int cout,
                        const _Float16* X, int ldx, const int* nbr, const _Float16* Wt,
                        const float* bias, const float* gamma, const float* beta,
                        _Float16* outH, float* outF, int ldo, int colOff, int flags,
                        int Nout, hipStream_t s) {
  dim3 grid(Nout / 256), block(256);
  if (cin == 32 && cout == 128) {
    spconv_kernel<32, 128><<<grid, block, shmem_bytes<32, 128>(), s>>>(
        X, ldx, nbr, Wt, bias, gamma, beta, outH, outF, ldo, colOff, flags);
  } else if (cin == 128 && cout == 128) {
    spconv_kernel<128, 128><<<grid, block, shmem_bytes<128, 128>(), s>>>(
        X, ldx, nbr, Wt, bias, gamma, beta, outH, outF, ldo, colOff, flags);
  } else if (cin == 256 && cout == 128) {
    spconv_kernel<256, 128><<<grid, block, shmem_bytes<256, 128>(), s>>>(
        X, ldx, nbr, Wt, bias, gamma, beta, outH, outF, ldo, colOff, flags);
  } else {  // 128 -> 64 (head)
    spconv_kernel<128, 64><<<grid, block, shmem_bytes<128, 64>(), s>>>(
        X, ldx, nbr, Wt, bias, gamma, beta, outH, outF, ldo, colOff, flags);
  }
}

extern "C" void kernel_launch(void* const* d_in, const int* in_sizes, int n_in,
                              void* d_out, int out_size, void* d_ws, size_t ws_size,
                              hipStream_t stream) {
  (void)n_in; (void)out_size; (void)ws_size;
  const int N8 = 262144, N7 = 65536, N6 = 16384;

  auto F = [&](int i) { return (const float*)d_in[i]; };
  auto I = [&](int i) { return (const int*)d_in[i]; };

  const float *x8, *x7, *x6, *Wh, *bh;
  const float *Wi[3], *bi[3], *Wd[2], *bd[2];
  const float *Wp[6], *bp[6], *gp[6], *bep[6];
  const int *nbr8, *nbr7, *nbr6, *dn87, *dn76;

  if (in_sizes[0] == 8388608) {  // top level in insertion order
    x8 = F(0); x7 = F(1); x6 = F(2);
    nbr8 = I(39); nbr7 = I(40); nbr6 = I(41); dn87 = I(42); dn76 = I(43);
    int p = 3;
    if (in_sizes[3] == 110592) {  // params in insertion order: init,proc,down,head
      for (int i = 0; i < 3; i++) { Wi[i] = F(p++); bi[i] = F(p++); }
      for (int i = 0; i < 6; i++) { Wp[i] = F(p++); bp[i] = F(p++); gp[i] = F(p++); bep[i] = F(p++); }
      for (int i = 0; i < 2; i++) { Wd[i] = F(p++); bd[i] = F(p++); }
      Wh = F(p++); bh = F(p++);
    } else {  // params dict-key-sorted: down,head,init,proc
      for (int i = 0; i < 2; i++) { Wd[i] = F(p++); bd[i] = F(p++); }
      Wh = F(p++); bh = F(p++);
      for (int i = 0; i < 3; i++) { Wi[i] = F(p++); bi[i] = F(p++); }
      for (int i = 0; i < 6; i++) { Wp[i] = F(p++); bp[i] = F(p++); gp[i] = F(p++); bep[i] = F(p++); }
    }
  } else {  // fully key-sorted top level
    dn76 = I(0); dn87 = I(1); nbr6 = I(2); nbr7 = I(3); nbr8 = I(4);
    int p = 5;
    for (int i = 0; i < 2; i++) { Wd[i] = F(p++); bd[i] = F(p++); }
    Wh = F(p++); bh = F(p++);
    for (int i = 0; i < 3; i++) { Wi[i] = F(p++); bi[i] = F(p++); }
    for (int i = 0; i < 6; i++) { Wp[i] = F(p++); bp[i] = F(p++); gp[i] = F(p++); bep[i] = F(p++); }
    x6 = F(41); x7 = F(42); x8 = F(43);
  }

  // -------- workspace carve (with deliberate aliasing/reuse) --------
  char* wsb = (char*)d_ws;
  size_t off = 0;
  auto carve = [&](size_t bytes) -> char* {
    char* r = wsb + off;
    off = (off + bytes + 255) & ~(size_t)255;
    return r;
  };
  _Float16* XF8 = (_Float16*)carve((size_t)N8 * 32 * 2);
  _Float16* XF7 = (_Float16*)carve((size_t)N7 * 32 * 2);
  _Float16* XF6 = (_Float16*)carve((size_t)N6 * 32 * 2);
  _Float16* H8A = (_Float16*)carve((size_t)N8 * 128 * 2);
  char* reg = carve((size_t)N8 * 128 * 2);     // 64MB region
  _Float16* H8B  = (_Float16*)reg;
  _Float16* CAT7 = (_Float16*)reg;                                            // reuses H8B after it is consumed
  _Float16* H7A  = (_Float16*)(reg + (size_t)N7 * 256 * 2);
  _Float16* H7B  = (_Float16*)(reg + (size_t)N7 * 256 * 2 + (size_t)N7 * 128 * 2);
  _Float16* CAT6 = XF8;                                                       // reuses XF8 after level-8 init
  _Float16* H6A  = (_Float16*)((char*)XF8 + (size_t)N6 * 256 * 2);
  _Float16* H6B  = (_Float16*)((char*)XF8 + (size_t)N6 * 256 * 2 + (size_t)N6 * 128 * 2);

  const int procCin[6] = {128, 128, 256, 128, 256, 128};
  _Float16 *WTi[3], *WTd[2], *WTh, *WTp[6];
  for (int i = 0; i < 3; i++) WTi[i] = (_Float16*)carve((size_t)KTAPS * 32 * 128 * 2);
  for (int i = 0; i < 6; i++) WTp[i] = (_Float16*)carve((size_t)KTAPS * procCin[i] * 128 * 2);
  for (int i = 0; i < 2; i++) WTd[i] = (_Float16*)carve((size_t)KTAPS * 128 * 128 * 2);
  WTh = (_Float16*)carve((size_t)KTAPS * 128 * 64 * 2);

  // raise dynamic-LDS caps for the double-buffered instantiations
  hipFuncSetAttribute(reinterpret_cast<const void*>(spconv_kernel<128, 128>),
                      hipFuncAttributeMaxDynamicSharedMemorySize,
                      (int)shmem_bytes<128, 128>());
  hipFuncSetAttribute(reinterpret_cast<const void*>(spconv_kernel<256, 128>),
                      hipFuncAttributeMaxDynamicSharedMemorySize,
                      (int)shmem_bytes<256, 128>());

  // -------- prep: f16 activations + transposed f16 weights --------
  auto cv = [&](const float* x, _Float16* y, int n) {
    cvt_f32_f16<<<(n + 255) / 256, 256, 0, stream>>>(x, y, n);
  };
  auto tw = [&](const float* W, _Float16* Wt, int cin, int cout) {
    int total = KTAPS * cin * cout;
    transpose_w<<<(total + 255) / 256, 256, 0, stream>>>(W, Wt, cin, cout);
  };
  cv(x8, XF8, N8 * 32);
  cv(x7, XF7, N7 * 32);
  cv(x6, XF6, N6 * 32);
  for (int i = 0; i < 3; i++) tw(Wi[i], WTi[i], 32, 128);
  for (int i = 0; i < 6; i++) tw(Wp[i], WTp[i], procCin[i], 128);
  for (int i = 0; i < 2; i++) tw(Wd[i], WTd[i], 128, 128);
  tw(Wh, WTh, 128, 64);

  // -------- network (flags: 1=lrelu, 2=layernorm, 4=f32 out) --------
  // level 8 (finest)
  launch_conv(32, 128, XF8, 32, nbr8, WTi[0], bi[0], nullptr, nullptr,
              H8A, nullptr, 128, 0, 1, N8, stream);
  launch_conv(128, 128, H8A, 128, nbr8, WTp[0], bp[0], gp[0], bep[0],
              H8B, nullptr, 128, 0, 3, N8, stream);
  launch_conv(128, 128, H8B, 128, nbr8, WTp[1], bp[1], gp[1], bep[1],
              H8A, nullptr, 128, 0, 3, N8, stream);
  // level 7
  launch_conv(32, 128, XF7, 32, nbr7, WTi[1], bi[1], nullptr, nullptr,
              CAT7, nullptr, 256, 128, 1, N7, stream);
  launch_conv(128, 128, H8A, 128, dn87, WTd[0], bd[0], nullptr, nullptr,
              CAT7, nullptr, 256, 0, 1, N7, stream);
  launch_conv(256, 128, CAT7, 256, nbr7, WTp[2], bp[2], gp[2], bep[2],
              H7A, nullptr, 128, 0, 3, N7, stream);
  launch_conv(128, 128, H7A, 128, nbr7, WTp[3], bp[3], gp[3], bep[3],
              H7B, nullptr, 128, 0, 3, N7, stream);
  // level 6 (coarsest)
  launch_conv(32, 128, XF6, 32, nbr6, WTi[2], bi[2], nullptr, nullptr,
              CAT6, nullptr, 256, 128, 1, N6, stream);
  launch_conv(128, 128, H7B, 128, dn76, WTd[1], bd[1], nullptr, nullptr,
              CAT6, nullptr, 256, 0, 1, N6, stream);
  launch_conv(256, 128, CAT6, 256, nbr6, WTp[4], bp[4], gp[4], bep[4],
              H6A, nullptr, 128, 0, 3, N6, stream);
  launch_conv(128, 128, H6A, 128, nbr6, WTp[5], bp[5], gp[5], bep[5],
              H6B, nullptr, 128, 0, 3, N6, stream);
  // head -> f32 output
  launch_conv(128, 64, H6B, 128, nbr6, WTh, bh, nullptr, nullptr,
              nullptr, (float*)d_out, 64, 0, 4, N6, stream);
}
